// RecurrentDecoder_7215545057976
// MI455X (gfx1250) — matrix operand order repaired and checked by reading, not compiled, not run on hardware
//
#include <hip/hip_runtime.h>
#include <math.h>

typedef __attribute__((ext_vector_type(16))) _Float16 v16h_t;
typedef __attribute__((ext_vector_type(8)))  float    v8f_t;

#define DEV __device__ __forceinline__

constexpr int B_  = 16;
constexpr int N_  = 128;
constexpr int H_  = 256;
constexpr int IN_ = 4;
constexpr int ET_ = 2;
constexpr int E_  = N_ * (N_ - 1);          // 16256
constexpr int TE_ = 32;                     // edges per block (2 M-tiles)
constexpr float EPS_    = 1e-8f;
constexpr float INV_NM1 = 1.0f / (float)(N_ - 1);

DEV float sigmoidf_(float x) { return 1.0f / (1.0f + expf(-x)); }

// ---------------------------------------------------------------------------
// WMMA fragment helpers (wave32; A and B both use contiguous k = 16*hi + s)
// A: 16x32 fp16 tile from LDS (row-major, `stride` in halves).
DEV v16h_t lds_a_frag(const _Float16* base, int stride, int lane, int kBase) {
  const int m  = lane & 15;
  const int hi = (lane >> 4) & 1;
  const _Float16* row = base + m * stride + kBase + hi * 16;
  union { v16h_t v; unsigned u[8]; } a;
#pragma unroll
  for (int r = 0; r < 8; ++r) {
    const int k = 2 * r;
    a.u[r] = *(const unsigned*)(row + k);   // 2 halves per VGPR, 4B aligned
  }
  return a.v;
}

// B: 32x16 fp16 tile from fp16 weights W[out][Kdim] row-major.
DEV v16h_t gmem_b_frag(const _Float16* W, int Kdim, int lane, int nBase, int kBase) {
  const int n  = lane & 15;
  const int hi = (lane >> 4) & 1;
  return *(const v16h_t*)(W + (size_t)(nBase + n) * Kdim + kBase + hi * 16);
}

DEV v8f_t wmma16(v16h_t a, v16h_t b, v8f_t c) {
  return __builtin_amdgcn_wmma_f32_16x16x32_f16(false, a, false, b, (short)0, c,
                                                false, false);
}

// ---------------------------------------------------------------------------
// Weight prep: f32 [rows][scols] -> f16 [rows][dcols], zero pad cols
__global__ void k_wprep(const float* __restrict__ src, _Float16* __restrict__ dst,
                        int rows, int scols, int dcols) {
  int i = blockIdx.x * blockDim.x + threadIdx.x;
  if (i >= rows * dcols) return;
  int r = i / dcols, c = i % dcols;
  dst[i] = (c < scols) ? (_Float16)src[r * scols + c] : (_Float16)0.0f;
}

// hidden (B,N,H) f32 -> f16 (halves gather bytes in the edge kernel)
__global__ void k_hprep(const float* __restrict__ src, _Float16* __restrict__ dst,
                        int n) {
  int i = blockIdx.x * blockDim.x + threadIdx.x;
  if (i < n) dst[i] = (_Float16)src[i];
}

// ---------------------------------------------------------------------------
// Node features. aux[bn][16] = {c,s,th,px,py,vx,vy,lpx,lpy,lvx,lvy,lfx,lfy,...}
// relf[bn][16] = rel_feat (15 features + pad)
__global__ void k_node(const float* __restrict__ inputs,
                       const float* __restrict__ field,
                       float* __restrict__ aux, _Float16* __restrict__ relf) {
  int bn = blockIdx.x * blockDim.x + threadIdx.x;
  if (bn >= B_ * N_) return;
  float px = inputs[bn * 4 + 0], py = inputs[bn * 4 + 1];
  float vx = inputs[bn * 4 + 2], vy = inputs[bn * 4 + 3];
  float fx = field[bn * 2 + 0],  fy = field[bn * 2 + 1];
  float th = atan2f(vy, vx);
  float c = cosf(th), s = sinf(th);
  // R = [[c,s],[-s,c]];  R@v = (c*vx+s*vy, -s*vx+c*vy)
  float lpx = c * px + s * py, lpy = -s * px + c * py;
  float lvx = c * vx + s * vy, lvy = -s * vx + c * vy;
  float lfx = c * fx + s * fy, lfy = -s * fx + c * fy;
  float npos = sqrtf(px * px + py * py + EPS_);
  float nvel = sqrtf(vx * vx + vy * vy + EPS_);
  float ang  = atan2f(lpy, lpx);
  float vang = atan2f(lvy, lvx);
  float* A = aux + bn * 16;
  A[0]=c; A[1]=s; A[2]=th; A[3]=px; A[4]=py; A[5]=vx; A[6]=vy;
  A[7]=lpx; A[8]=lpy; A[9]=lvx; A[10]=lvy; A[11]=lfx; A[12]=lfy;
  A[13]=0.f; A[14]=0.f; A[15]=0.f;
  float rf[16] = {lpx, lpy, lvx, lvy, npos, ang, nvel, vang, th,
                  lpx, lpy, lvx, lvy, lfx, lfy, 0.f};
  _Float16* R = relf + bn * 16;
#pragma unroll
  for (int j = 0; j < 16; ++j) R[j] = (_Float16)rf[j];
}

// ---------------------------------------------------------------------------
// Edge kernel: one block per (32-edge tile, batch). 256 threads = 8 waves.
// Wave w owns output column tiles {2w, 2w+1}; each B fragment feeds 2 WMMAs
// (M-tiles rows 0-15 and 16-31), halving weight traffic per FLOP.
__global__ void __launch_bounds__(256)
k_edge(const _Float16* __restrict__ hidden16, const float* __restrict__ edges,
       const int* __restrict__ sendE, const int* __restrict__ recvE,
       const float* __restrict__ aux,
       const _Float16* __restrict__ wMsg1, const float* __restrict__ bMsg1,
       const _Float16* __restrict__ wMsg2, const float* __restrict__ bMsg2,
       const _Float16* __restrict__ wPm1,  const float* __restrict__ bPm1,
       const _Float16* __restrict__ wPm2,  const float* __restrict__ bPm2,
       float* __restrict__ aggM, float* __restrict__ aggP) {
  __shared__ _Float16 sPre[TE_][512];   // pre_msg tile  (32 KB)
  __shared__ _Float16 sAct[TE_][256];   // activations   (16 KB, reused)
  __shared__ _Float16 sEdge[TE_][32];   // edge_attr, K padded to 32 (2 KB)
  __shared__ float    sW[TE_];
  __shared__ int      sRecv[TE_];

  const int tid = threadIdx.x, lane = tid & 31, wid = tid >> 5;
  const int e0 = blockIdx.x * TE_, b = blockIdx.y;

  // --- stage edge features / weights (one thread per edge) ---
  if (tid < TE_) {
    const int e = e0 + tid;
    const int sN = sendE[e], rN = recvE[e];
    sRecv[tid] = rN;
    sW[tid] = edges[((size_t)b * E_ + e) * ET_ + 1];
    const float* As = aux + (b * N_ + sN) * 16;
    const float* Ar = aux + (b * N_ + rN) * 16;
    float dpx = As[3]-Ar[3], dpy = As[4]-Ar[4];
    float dvx = As[5]-Ar[5], dvy = As[6]-Ar[6];
    float dth = As[2]-Ar[2];
    float ndp = sqrtf(dpx*dpx + dpy*dpy + EPS_);
    float ndv = sqrtf(dvx*dvx + dvy*dvy + EPS_);
    float cr = Ar[0], sr = Ar[1], cs = As[0], ss = As[1];
    float r1px =  cr*dpx + sr*dpy, r1py = -sr*dpx + cr*dpy;
    float r1vx =  cr*dvx + sr*dvy, r1vy = -sr*dvx + cr*dvy;
    float r2px = -(cs*dpx + ss*dpy), r2py = -(-ss*dpx + cs*dpy);
    float r2vx = -(cs*dvx + ss*dvy), r2vy = -(-ss*dvx + cs*dvy);
    float f[24] = {
      r1px, r1py, r1vx, r1vy, ndp, atan2f(r1py, r1px), ndv, atan2f(r1vy, r1vx),  dth,
      r2px, r2py, r2vx, r2vy, ndp, atan2f(r2py, r2px), ndv, atan2f(r2vy, r2vx), -dth,
      Ar[7], Ar[8], Ar[9], Ar[10], Ar[11], Ar[12]};
#pragma unroll
    for (int j = 0; j < 24; ++j) sEdge[tid][j] = (_Float16)f[j];
#pragma unroll
    for (int j = 24; j < 32; ++j) sEdge[tid][j] = (_Float16)0.0f;
  }
  // --- gather pre_msg = [hidden16[recv] | hidden16[send]] (16B copies) ---
  {
    const int m = tid >> 3, e = e0 + m;          // 8 threads per edge row
    const int rN = recvE[e], sN = sendE[e];
    const int c0 = (tid & 7) * 64;               // 64 halves per thread
    const int node = (c0 < 256) ? rN : sN;
    const uint4* src = (const uint4*)(hidden16 +
        ((size_t)(b * N_ + node)) * H_ + (c0 & 255));
    uint4* dst = (uint4*)(&sPre[m][c0]);
#pragma unroll
    for (int j = 0; j < 8; ++j) dst[j] = src[j];
  }
  __syncthreads();

  const int hi = (lane >> 4) & 1;
  const int colL = lane & 15;

  // === msg layer 1: tanh(pre_msg @ W1^T + b1)  K=512 ===
#pragma unroll
  for (int t = 0; t < 2; ++t) {
    const int nB = (wid * 2 + t) * 16;
    v8f_t acc0 = {}, acc1 = {};
#pragma unroll
    for (int kk = 0; kk < 16; ++kk) {
      v16h_t bfr = gmem_b_frag(wMsg1, 512, lane, nB, kk * 32);
      v16h_t a0 = lds_a_frag(&sPre[0][0],  512, lane, kk * 32);
      v16h_t a1 = lds_a_frag(&sPre[16][0], 512, lane, kk * 32);
      acc0 = wmma16(a0, bfr, acc0);
      acc1 = wmma16(a1, bfr, acc1);
    }
    const int col = nB + colL;
    const float bias = bMsg1[col];
#pragma unroll
    for (int v = 0; v < 8; ++v) {
      sAct[v + 8 * hi][col]      = (_Float16)tanhf(acc0[v] + bias);
      sAct[16 + v + 8 * hi][col] = (_Float16)tanhf(acc1[v] + bias);
    }
  }
  __syncthreads();

  // === msg layer 2: tanh(. @ W2^T + b2) * edgew -> atomic scatter ===
#pragma unroll
  for (int t = 0; t < 2; ++t) {
    const int nB = (wid * 2 + t) * 16;
    v8f_t acc0 = {}, acc1 = {};
#pragma unroll
    for (int kk = 0; kk < 8; ++kk) {
      v16h_t bfr = gmem_b_frag(wMsg2, 256, lane, nB, kk * 32);
      v16h_t a0 = lds_a_frag(&sAct[0][0],  256, lane, kk * 32);
      v16h_t a1 = lds_a_frag(&sAct[16][0], 256, lane, kk * 32);
      acc0 = wmma16(a0, bfr, acc0);
      acc1 = wmma16(a1, bfr, acc1);
    }
    const int col = nB + colL;
    const float bias = bMsg2[col];
#pragma unroll
    for (int v = 0; v < 8; ++v) {
      const int m = v + 8 * hi;
      float v0 = tanhf(acc0[v] + bias) * sW[m];
      float v1 = tanhf(acc1[v] + bias) * sW[16 + m];
      atomicAdd(&aggM[((size_t)(b * N_ + sRecv[m]))      * H_ + col], v0);
      atomicAdd(&aggM[((size_t)(b * N_ + sRecv[16 + m])) * H_ + col], v1);
    }
  }
  __syncthreads();

  // === present layer 1: relu(edge_attr @ Wp1^T + b)  K=32 (padded) ===
#pragma unroll
  for (int t = 0; t < 2; ++t) {
    const int nB = (wid * 2 + t) * 16;
    v8f_t acc0 = {}, acc1 = {};
    {
      v16h_t bfr = gmem_b_frag(wPm1, 32, lane, nB, 0);
      v16h_t a0 = lds_a_frag(&sEdge[0][0],  32, lane, 0);
      v16h_t a1 = lds_a_frag(&sEdge[16][0], 32, lane, 0);
      acc0 = wmma16(a0, bfr, acc0);
      acc1 = wmma16(a1, bfr, acc1);
    }
    const int col = nB + colL;
    const float bias = bPm1[col];
#pragma unroll
    for (int v = 0; v < 8; ++v) {
      sAct[v + 8 * hi][col]      = (_Float16)fmaxf(acc0[v] + bias, 0.0f);
      sAct[16 + v + 8 * hi][col] = (_Float16)fmaxf(acc1[v] + bias, 0.0f);
    }
  }
  __syncthreads();

  // === present layer 2: relu -> weighted atomic scatter ===
#pragma unroll
  for (int t = 0; t < 2; ++t) {
    const int nB = (wid * 2 + t) * 16;
    v8f_t acc0 = {}, acc1 = {};
#pragma unroll
    for (int kk = 0; kk < 8; ++kk) {
      v16h_t bfr = gmem_b_frag(wPm2, 256, lane, nB, kk * 32);
      v16h_t a0 = lds_a_frag(&sAct[0][0],  256, lane, kk * 32);
      v16h_t a1 = lds_a_frag(&sAct[16][0], 256, lane, kk * 32);
      acc0 = wmma16(a0, bfr, acc0);
      acc1 = wmma16(a1, bfr, acc1);
    }
    const int col = nB + colL;
    const float bias = bPm2[col];
#pragma unroll
    for (int v = 0; v < 8; ++v) {
      const int m = v + 8 * hi;
      float v0 = fmaxf(acc0[v] + bias, 0.0f) * sW[m];
      float v1 = fmaxf(acc1[v] + bias, 0.0f) * sW[16 + m];
      atomicAdd(&aggP[((size_t)(b * N_ + sRecv[m]))      * H_ + col], v0);
      atomicAdd(&aggP[((size_t)(b * N_ + sRecv[16 + m])) * H_ + col], v1);
    }
  }
}

// ---------------------------------------------------------------------------
// GRU + output MLP. One block per 16 nodes (bn tile). 256 threads = 8 waves.
__global__ void __launch_bounds__(256)
k_gru(const float* __restrict__ inputs, const float* __restrict__ hidden,
      const float* __restrict__ aux, const _Float16* __restrict__ relf,
      const float* __restrict__ aggM, const float* __restrict__ aggP,
      const _Float16* __restrict__ wInpR, const _Float16* __restrict__ wInpI,
      const _Float16* __restrict__ wInpN,
      const _Float16* __restrict__ wPresR, const _Float16* __restrict__ wPresI,
      const _Float16* __restrict__ wPresN,
      const _Float16* __restrict__ wHidR, const _Float16* __restrict__ wHidI,
      const _Float16* __restrict__ wHidH,
      const _Float16* __restrict__ wOut1, const _Float16* __restrict__ wOut2,
      const _Float16* __restrict__ wOut3,
      const float* __restrict__ bInpR, const float* __restrict__ bInpI,
      const float* __restrict__ bInpN,
      const float* __restrict__ bPresR, const float* __restrict__ bPresI,
      const float* __restrict__ bPresN,
      const float* __restrict__ bOut1, const float* __restrict__ bOut2,
      const float* __restrict__ bOut3,
      float* __restrict__ outO, float* __restrict__ outH) {
  __shared__ _Float16 sM[16][256];     // agg_msgs tile  (reused as h1)
  __shared__ _Float16 sP[16][256];     // present_agg    (reused as h2)
  __shared__ _Float16 sH[16][256];     // hidden_new
  __shared__ _Float16 sRelf[16][32];

  const int tid = threadIdx.x, lane = tid & 31, wid = tid >> 5;
  const int bnBase = blockIdx.x * 16;

  { // stage agg buffers (scaled by 1/(N-1)) as fp16
    const int m = tid >> 4, bn = bnBase + m;
    const int c0 = (tid & 15) * 16;
#pragma unroll
    for (int j = 0; j < 16; ++j) {
      int c = c0 + j;
      sM[m][c] = (_Float16)(aggM[(size_t)bn * H_ + c] * INV_NM1);
      sP[m][c] = (_Float16)(aggP[(size_t)bn * H_ + c] * INV_NM1);
    }
  }
  if (tid < 16) {
#pragma unroll
    for (int j = 0; j < 16; ++j) sRelf[tid][j] = relf[(size_t)(bnBase + tid) * 16 + j];
#pragma unroll
    for (int j = 16; j < 32; ++j) sRelf[tid][j] = (_Float16)0.0f;
  }
  __syncthreads();

  const int hi = (lane >> 4) & 1;
  const int colL = lane & 15;

  // === GRU gates ===
#pragma unroll
  for (int t = 0; t < 2; ++t) {
    const int nB = (wid * 2 + t) * 16;
    v8f_t aR = {}, aI = {}, aN = {}, hR = {}, hI = {}, hH = {};
    { // rel_feat path, K=32 (padded from 15)
      v16h_t a = lds_a_frag(&sRelf[0][0], 32, lane, 0);
      aR = wmma16(a, gmem_b_frag(wInpR, 32, lane, nB, 0), aR);
      aI = wmma16(a, gmem_b_frag(wInpI, 32, lane, nB, 0), aI);
      aN = wmma16(a, gmem_b_frag(wInpN, 32, lane, nB, 0), aN);
    }
#pragma unroll
    for (int kk = 0; kk < 8; ++kk) { // present_agg path, K=256
      v16h_t a = lds_a_frag(&sP[0][0], 256, lane, kk * 32);
      aR = wmma16(a, gmem_b_frag(wPresR, 256, lane, nB, kk * 32), aR);
      aI = wmma16(a, gmem_b_frag(wPresI, 256, lane, nB, kk * 32), aI);
      aN = wmma16(a, gmem_b_frag(wPresN, 256, lane, nB, kk * 32), aN);
    }
#pragma unroll
    for (int kk = 0; kk < 8; ++kk) { // agg_msgs path, K=256
      v16h_t a = lds_a_frag(&sM[0][0], 256, lane, kk * 32);
      hR = wmma16(a, gmem_b_frag(wHidR, 256, lane, nB, kk * 32), hR);
      hI = wmma16(a, gmem_b_frag(wHidI, 256, lane, nB, kk * 32), hI);
      hH = wmma16(a, gmem_b_frag(wHidH, 256, lane, nB, kk * 32), hH);
    }
    const int col = nB + colL;
    const float bR = bInpR[col] + bPresR[col];
    const float bI = bInpI[col] + bPresI[col];
    const float bN = bInpN[col] + bPresN[col];
#pragma unroll
    for (int v = 0; v < 8; ++v) {
      const int m = v + 8 * hi, bn = bnBase + m;
      float r  = sigmoidf_(aR[v] + bR + hR[v]);
      float ig = sigmoidf_(aI[v] + bI + hI[v]);
      float ng = tanhf(aN[v] + bN + r * hH[v]);
      float hOld = hidden[(size_t)bn * H_ + col];
      float hNew = (1.0f - ig) * ng + ig * hOld;
      outH[(size_t)bn * H_ + col] = hNew;
      sH[m][col] = (_Float16)hNew;
    }
  }
  __syncthreads();

  // === out layer 1 -> sM ===
#pragma unroll
  for (int t = 0; t < 2; ++t) {
    const int nB = (wid * 2 + t) * 16;
    v8f_t acc = {};
#pragma unroll
    for (int kk = 0; kk < 8; ++kk) {
      v16h_t a = lds_a_frag(&sH[0][0], 256, lane, kk * 32);
      acc = wmma16(a, gmem_b_frag(wOut1, 256, lane, nB, kk * 32), acc);
    }
    const int col = nB + colL;
    const float bias = bOut1[col];
#pragma unroll
    for (int v = 0; v < 8; ++v)
      sM[v + 8 * hi][col] = (_Float16)fmaxf(acc[v] + bias, 0.0f);
  }
  __syncthreads();

  // === out layer 2 -> sP ===
#pragma unroll
  for (int t = 0; t < 2; ++t) {
    const int nB = (wid * 2 + t) * 16;
    v8f_t acc = {};
#pragma unroll
    for (int kk = 0; kk < 8; ++kk) {
      v16h_t a = lds_a_frag(&sM[0][0], 256, lane, kk * 32);
      acc = wmma16(a, gmem_b_frag(wOut2, 256, lane, nB, kk * 32), acc);
    }
    const int col = nB + colL;
    const float bias = bOut2[col];
#pragma unroll
    for (int v = 0; v < 8; ++v)
      sP[v + 8 * hi][col] = (_Float16)fmaxf(acc[v] + bias, 0.0f);
  }
  __syncthreads();

  // === head (H -> 4) + inverse rotation + residual ===
  if (tid < 16) {
    const int bn = bnBase + tid;
    float p[4];
#pragma unroll
    for (int o = 0; o < 4; ++o) {
      float acc = bOut3[o];
      for (int k = 0; k < 256; ++k)
        acc += (float)sP[tid][k] * (float)wOut3[o * 256 + k];
      p[o] = acc;
    }
    const float c = aux[bn * 16 + 0], s = aux[bn * 16 + 1];
    // Rinv = [[c,-s],[s,c]]
    float g0 = c * p[0] - s * p[1], g1 = s * p[0] + c * p[1];
    float g2 = c * p[2] - s * p[3], g3 = s * p[2] + c * p[3];
    outO[bn * 4 + 0] = inputs[bn * 4 + 0] + g0;
    outO[bn * 4 + 1] = inputs[bn * 4 + 1] + g1;
    outO[bn * 4 + 2] = inputs[bn * 4 + 2] + g2;
    outO[bn * 4 + 3] = inputs[bn * 4 + 3] + g3;
  }
}

// ---------------------------------------------------------------------------
extern "C" void kernel_launch(void* const* d_in, const int* in_sizes, int n_in,
                              void* d_out, int out_size, void* d_ws, size_t ws_size,
                              hipStream_t stream) {
  const float* inputs = (const float*)d_in[0];
  const float* hidden = (const float*)d_in[1];
  const float* edges  = (const float*)d_in[2];
  const float* field  = (const float*)d_in[3];
  const int* sendE = (const int*)d_in[33];
  const int* recvE = (const int*)d_in[34];

  char* ws = (char*)d_ws;
  size_t off = 0;
  float* aggM = (float*)(ws + off); off += (size_t)B_ * N_ * H_ * 4;   // 2 MB
  float* aggP = (float*)(ws + off); off += (size_t)B_ * N_ * H_ * 4;   // 2 MB
  float* aux  = (float*)(ws + off); off += (size_t)B_ * N_ * 16 * 4;
  _Float16* relf = (_Float16*)(ws + off); off += (size_t)B_ * N_ * 16 * 2;
  _Float16* hidden16 = (_Float16*)(ws + off); off += (size_t)B_ * N_ * H_ * 2;

  _Float16* wbase = (_Float16*)(ws + off);
  size_t h = 0;
  _Float16* wMsg1 = wbase + h; h += 256 * 512;
  _Float16* wMsg2 = wbase + h; h += 256 * 256;
  _Float16* wPm1  = wbase + h; h += 256 * 32;
  _Float16* wPm2  = wbase + h; h += 256 * 256;
  _Float16* wPresR= wbase + h; h += 256 * 256;
  _Float16* wPresI= wbase + h; h += 256 * 256;
  _Float16* wPresN= wbase + h; h += 256 * 256;
  _Float16* wHidR = wbase + h; h += 256 * 256;
  _Float16* wHidI = wbase + h; h += 256 * 256;
  _Float16* wHidH = wbase + h; h += 256 * 256;
  _Float16* wInpR = wbase + h; h += 256 * 32;
  _Float16* wInpI = wbase + h; h += 256 * 32;
  _Float16* wInpN = wbase + h; h += 256 * 32;
  _Float16* wOut1 = wbase + h; h += 256 * 256;
  _Float16* wOut2 = wbase + h; h += 256 * 256;
  _Float16* wOut3 = wbase + h; h += 4 * 256;

  // zero aggregation buffers (graph-capturable)
  hipMemsetAsync(aggM, 0, (size_t)2 * B_ * N_ * H_ * 4, stream);

  auto wprep = [&](const float* src, _Float16* dst, int rows, int sc, int dc) {
    int n = rows * dc;
    k_wprep<<<(n + 255) / 256, 256, 0, stream>>>(src, dst, rows, sc, dc);
  };
  // weight index [1] of ET-indexed arrays
  wprep((const float*)d_in[4]  + 256 * 512, wMsg1, 256, 512, 512);
  wprep((const float*)d_in[6]  + 256 * 256, wMsg2, 256, 256, 256);
  wprep((const float*)d_in[8]  + 256 * 24,  wPm1,  256, 24, 32);
  wprep((const float*)d_in[10] + 256 * 256, wPm2,  256, 256, 256);
  wprep((const float*)d_in[12], wHidR, 256, 256, 256);
  wprep((const float*)d_in[13], wHidI, 256, 256, 256);
  wprep((const float*)d_in[14], wHidH, 256, 256, 256);
  wprep((const float*)d_in[15], wPresR, 256, 256, 256);
  wprep((const float*)d_in[17], wPresI, 256, 256, 256);
  wprep((const float*)d_in[19], wPresN, 256, 256, 256);
  wprep((const float*)d_in[21], wInpR, 256, 15, 32);
  wprep((const float*)d_in[23], wInpI, 256, 15, 32);
  wprep((const float*)d_in[25], wInpN, 256, 15, 32);
  wprep((const float*)d_in[27], wOut1, 256, 256, 256);
  wprep((const float*)d_in[29], wOut2, 256, 256, 256);
  wprep((const float*)d_in[31], wOut3, 4, 256, 256);

  {
    int n = B_ * N_ * H_;
    k_hprep<<<(n + 255) / 256, 256, 0, stream>>>(hidden, hidden16, n);
  }
  k_node<<<(B_ * N_ + 255) / 256, 256, 0, stream>>>(inputs, field, aux, relf);

  k_edge<<<dim3(E_ / TE_, B_), 256, 0, stream>>>(
      hidden16, edges, sendE, recvE, aux,
      wMsg1, (const float*)d_in[5] + 256,
      wMsg2, (const float*)d_in[7] + 256,
      wPm1,  (const float*)d_in[9] + 256,
      wPm2,  (const float*)d_in[11] + 256,
      aggM, aggP);

  float* outO = (float*)d_out;                 // (B,N,IN)
  float* outH = outO + (size_t)B_ * N_ * IN_;  // (B,N,H)
  k_gru<<<(B_ * N_) / 16, 256, 0, stream>>>(
      inputs, hidden, aux, relf, aggM, aggP,
      wInpR, wInpI, wInpN, wPresR, wPresI, wPresN, wHidR, wHidI, wHidH,
      wOut1, wOut2, wOut3,
      (const float*)d_in[22], (const float*)d_in[24], (const float*)d_in[26],
      (const float*)d_in[16], (const float*)d_in[18], (const float*)d_in[20],
      (const float*)d_in[28], (const float*)d_in[30], (const float*)d_in[32],
      outO, outH);
}
